// attention_10290741641600
// MI455X (gfx1250) — compile-verified, matrix operand order checked by reading, not run on hardware
//
#include <hip/hip_runtime.h>
#include <hip/hip_bf16.h>

// ---------------------------------------------------------------------------
// MI455X (gfx1250) multi-head attention, bf16 WMMA pipeline.
// All matmuls use v_wmma_f32_16x16x32_bf16 (wave32, 16x16 tiles, K=32).
// ---------------------------------------------------------------------------

typedef __attribute__((ext_vector_type(16))) __bf16 v16bf;
typedef __attribute__((ext_vector_type(8)))  __bf16 v8bf;
typedef __attribute__((ext_vector_type(4)))  __bf16 v4bf;
typedef __attribute__((ext_vector_type(8)))  float  v8f;

#define DIM_    768
#define NQKV    2304
#define SLEN    2048
#define NHEAD   12
#define HD      64
#define MROWS   8192          // B*S

static __device__ inline v8f vzero8() {
  v8f z = {0.f, 0.f, 0.f, 0.f, 0.f, 0.f, 0.f, 0.f};
  return z;
}

// Load one 16-lane-owned half-row fragment: 8 contiguous bf16 at p (K+0..7 or
// K+8..15 depending on lane half) and 8 at p+16 (K+16..23 / K+24..31).
// Matches the CDNA5 16-bit A-matrix 16x32 VGPR layout (and B, column-major).
static __device__ inline v16bf frag_ld(const __bf16* p) {
  v8bf lo = *(const v8bf*)p;
  v8bf hi = *(const v8bf*)(p + 16);
  v16bf r;
#pragma unroll
  for (int i = 0; i < 8; ++i) { r[i] = lo[i]; r[i + 8] = hi[i]; }
  return r;
}

static __device__ inline v8f wmma_bf16(v16bf a, v16bf b, v8f c) {
  // (neg_a, A, neg_b, B, c_mod, C, reuse_a, reuse_b)
  return __builtin_amdgcn_wmma_f32_16x16x32_bf16(false, a, false, b,
                                                 (short)0, c, false, false);
}

// ---------------------------------------------------------------------------
// Packing kernels: fp32 -> bf16, weights transposed to [N][K] so B-fragments
// are contiguous per lane.
// ---------------------------------------------------------------------------
__global__ void pack_x_kernel(const float* __restrict__ x,
                              __bf16* __restrict__ xb, int n) {
  int i = (blockIdx.x * blockDim.x + threadIdx.x) * 4;
  if (i >= n) return;
  float4 f = *(const float4*)(x + i);
  v4bf o;
  o[0] = (__bf16)f.x; o[1] = (__bf16)f.y; o[2] = (__bf16)f.z; o[3] = (__bf16)f.w;
  *(v4bf*)(xb + i) = o;
}

__global__ void pack_wqkv_kernel(const float* __restrict__ Wq,
                                 const float* __restrict__ Wk,
                                 const float* __restrict__ Wv,
                                 __bf16* __restrict__ Wt) {
  int i = blockIdx.x * blockDim.x + threadIdx.x;     // over NQKV*DIM_
  if (i >= NQKV * DIM_) return;
  int n = i / DIM_;
  int k = i - n * DIM_;
  float v;
  if (n < DIM_)             v = Wq[k * DIM_ + n];
  else if (n < 2 * DIM_)    v = Wk[k * DIM_ + (n - DIM_)];
  else                      v = Wv[k * DIM_ + (n - 2 * DIM_)];
  Wt[i] = (__bf16)v;
}

__global__ void pack_wo_kernel(const float* __restrict__ Wo,
                               __bf16* __restrict__ Wot) {
  int i = blockIdx.x * blockDim.x + threadIdx.x;     // over DIM_*DIM_
  if (i >= DIM_ * DIM_) return;
  int n = i / DIM_;
  int k = i - n * DIM_;
  Wot[i] = (__bf16)Wo[k * DIM_ + n];
}

// ---------------------------------------------------------------------------
// 64x64-per-wave register-blocked GEMM core: C += A[M,K] * Bt[N,K]^T.
// A, Bt row-major bf16; 16 WMMAs per K-step of 32.
// ---------------------------------------------------------------------------
static __device__ inline void gemm_acc_64x64(const __bf16* __restrict__ A,
                                             const __bf16* __restrict__ Bt,
                                             int K, int lda, int ldb,
                                             v8f acc[4][4], int lane16, int hi) {
  for (int k = 0; k < K; k += 32) {
    v16bf af[4], bfr[4];
#pragma unroll
    for (int i = 0; i < 4; ++i)
      af[i] = frag_ld(A + (size_t)(i * 16 + lane16) * lda + k + hi * 8);
#pragma unroll
    for (int j = 0; j < 4; ++j)
      bfr[j] = frag_ld(Bt + (size_t)(j * 16 + lane16) * ldb + k + hi * 8);
#pragma unroll
    for (int i = 0; i < 4; ++i)
#pragma unroll
      for (int j = 0; j < 4; ++j)
        acc[i][j] = wmma_bf16(af[i], bfr[j], acc[i][j]);
  }
}

// ---------------------------------------------------------------------------
// Fused QKV projection. Epilogue: bias, Q scaling (Hd^-0.5), head-major
// scatter; V stored transposed [B][H][64][S] so PV B-fragments are contiguous.
// ---------------------------------------------------------------------------
__global__ void __launch_bounds__(256)
qkv_gemm_kernel(const __bf16* __restrict__ xb, const __bf16* __restrict__ Wt,
                const float* __restrict__ bq, const float* __restrict__ bk,
                const float* __restrict__ bv,
                __bf16* __restrict__ Qb, __bf16* __restrict__ Kb,
                __bf16* __restrict__ Vt) {
  const int lane = threadIdx.x & 31;
  const int wave = threadIdx.x >> 5;
  const int tile = blockIdx.x * 8 + wave;     // 128 * 36 tiles
  const int tn = tile % (NQKV / 64);
  const int tm = tile / (NQKV / 64);
  const int m0 = tm * 64, n0 = tn * 64;
  const int lane16 = lane & 15, hi = lane >> 4;

  v8f acc[4][4];
#pragma unroll
  for (int i = 0; i < 4; ++i)
#pragma unroll
    for (int j = 0; j < 4; ++j) acc[i][j] = vzero8();

  gemm_acc_64x64(xb + (size_t)m0 * DIM_, Wt + (size_t)n0 * DIM_,
                 DIM_, DIM_, DIM_, acc, lane16, hi);

#pragma unroll
  for (int i = 0; i < 4; ++i) {
#pragma unroll
    for (int j = 0; j < 4; ++j) {
      const int ncol = n0 + j * 16 + lane16;
#pragma unroll
      for (int e = 0; e < 8; ++e) {
        const int m = m0 + i * 16 + e + hi * 8;
        const int b = m >> 11;            // / SLEN
        const int s = m & (SLEN - 1);
        float v = acc[i][j][e];
        if (ncol < DIM_) {                 // Q: bias + scale
          const int h = ncol >> 6, d = ncol & 63;
          v = (v + bq[ncol]) * 0.125f;     // 64^-0.5
          Qb[((size_t)((b * NHEAD + h) * SLEN + s)) * HD + d] = (__bf16)v;
        } else if (ncol < 2 * DIM_) {      // K
          const int nn = ncol - DIM_;
          const int h = nn >> 6, d = nn & 63;
          v += bk[nn];
          Kb[((size_t)((b * NHEAD + h) * SLEN + s)) * HD + d] = (__bf16)v;
        } else {                           // V, transposed
          const int nn = ncol - 2 * DIM_;
          const int h = nn >> 6, d = nn & 63;
          v += bv[nn];
          Vt[((size_t)((b * NHEAD + h) * HD + d)) * SLEN + s] = (__bf16)v;
        }
      }
    }
  }
}

// ---------------------------------------------------------------------------
// Flash attention: one wave per 16-query tile; 64-key tiles; online softmax.
// Probs round-trip through LDS to re-shape C-layout -> A-layout for PV WMMA.
// ---------------------------------------------------------------------------
__global__ void __launch_bounds__(256)
attn_kernel(const __bf16* __restrict__ Qb, const __bf16* __restrict__ Kb,
            const __bf16* __restrict__ Vt, const float* __restrict__ mask,
            __bf16* __restrict__ Ao) {
  __shared__ __bf16 lds_p[8][16 * 64];      // per-wave prob staging

  const int lane = threadIdx.x & 31;
  const int wave = threadIdx.x >> 5;
  const int tile = blockIdx.x * 8 + wave;   // B*H*(S/16) = 6144
  const int qt = tile & 127;                // S/16 = 128
  const int h  = (tile >> 7) % NHEAD;
  const int b  = tile / (128 * NHEAD);
  const int lane16 = lane & 15, hi = lane >> 4;
  const int q0 = qt * 16;

  const __bf16* Qbase = Qb + ((size_t)(b * NHEAD + h) * SLEN) * HD;
  const __bf16* Kbase = Kb + ((size_t)(b * NHEAD + h) * SLEN) * HD;
  const __bf16* Vbase = Vt + ((size_t)(b * NHEAD + h) * HD) * SLEN;
  const float*  mrow  = mask + (size_t)b * SLEN;

  // Q A-fragments (Hd = 64 -> two K-steps of 32)
  v16bf qf[2];
  qf[0] = frag_ld(Qbase + (size_t)(q0 + lane16) * HD + 0  + hi * 8);
  qf[1] = frag_ld(Qbase + (size_t)(q0 + lane16) * HD + 32 + hi * 8);

  v8f o[4];
#pragma unroll
  for (int nf = 0; nf < 4; ++nf) o[nf] = vzero8();
  float mstat[8], lstat[8];
#pragma unroll
  for (int e = 0; e < 8; ++e) { mstat[e] = -1e30f; lstat[e] = 0.f; }

  __bf16* pl = lds_p[wave];

  for (int kt = 0; kt < SLEN; kt += 64) {
    if (kt + 64 < SLEN)  // warm L2/L0 for next K tile
      __builtin_prefetch(Kbase + (size_t)(kt + 64 + lane) * HD, 0, 1);

    // ---- scores: 4 frags of 16 keys, QK^T over Hd=64 -------------------
    v8f s[4];
#pragma unroll
    for (int f = 0; f < 4; ++f) {
      const int key = kt + f * 16 + lane16;
      v16bf k0 = frag_ld(Kbase + (size_t)key * HD + 0  + hi * 8);
      v16bf k1 = frag_ld(Kbase + (size_t)key * HD + 32 + hi * 8);
      v8f c = vzero8();
      c = wmma_bf16(qf[0], k0, c);
      c = wmma_bf16(qf[1], k1, c);
      const float mv = mrow[key];          // additive mask over keys
#pragma unroll
      for (int e = 0; e < 8; ++e) c[e] += mv;
      s[f] = c;
    }

    // ---- online softmax (row stats live per lane-half) -----------------
    float rmax[8];
#pragma unroll
    for (int e = 0; e < 8; ++e)
      rmax[e] = fmaxf(fmaxf(s[0][e], s[1][e]), fmaxf(s[2][e], s[3][e]));
#pragma unroll
    for (int d = 1; d < 16; d <<= 1)
#pragma unroll
      for (int e = 0; e < 8; ++e)
        rmax[e] = fmaxf(rmax[e], __shfl_xor(rmax[e], d, 32));

    float corr[8];
#pragma unroll
    for (int e = 0; e < 8; ++e) {
      const float mnew = fmaxf(mstat[e], rmax[e]);
      corr[e] = __expf(mstat[e] - mnew);
      mstat[e] = mnew;
    }

    float rsum[8];
#pragma unroll
    for (int e = 0; e < 8; ++e) rsum[e] = 0.f;
#pragma unroll
    for (int f = 0; f < 4; ++f)
#pragma unroll
      for (int e = 0; e < 8; ++e) {
        const float p = __expf(s[f][e] - mstat[e]);
        s[f][e] = p;
        rsum[e] += p;
      }
#pragma unroll
    for (int d = 1; d < 16; d <<= 1)
#pragma unroll
      for (int e = 0; e < 8; ++e)
        rsum[e] += __shfl_xor(rsum[e], d, 32);
#pragma unroll
    for (int e = 0; e < 8; ++e) lstat[e] = lstat[e] * corr[e] + rsum[e];
#pragma unroll
    for (int nf = 0; nf < 4; ++nf)
#pragma unroll
      for (int e = 0; e < 8; ++e) o[nf][e] *= corr[e];

    // ---- probs C-layout -> LDS row-major [16 q][64 key] ----------------
#pragma unroll
    for (int f = 0; f < 4; ++f)
#pragma unroll
      for (int e = 0; e < 8; ++e)
        pl[(e + hi * 8) * 64 + f * 16 + lane16] = (__bf16)s[f][e];

    asm volatile("s_wait_dscnt 0x0" ::: "memory");

    v16bf p0 = frag_ld(pl + lane16 * 64 + 0  + hi * 8);   // keys kt+0..31
    v16bf p1 = frag_ld(pl + lane16 * 64 + 32 + hi * 8);   // keys kt+32..63

    // ---- PV: V^T rows are contiguous over keys -------------------------
#pragma unroll
    for (int nf = 0; nf < 4; ++nf) {
      const __bf16* vb = Vbase + (size_t)(nf * 16 + lane16) * SLEN + kt + hi * 8;
      v16bf v0 = frag_ld(vb);
      v16bf v1 = frag_ld(vb + 32);
      o[nf] = wmma_bf16(p0, v0, o[nf]);
      o[nf] = wmma_bf16(p1, v1, o[nf]);
    }
  }

  // ---- normalize & merge heads into [B*S][768] bf16 --------------------
  float inv[8];
#pragma unroll
  for (int e = 0; e < 8; ++e) inv[e] = 1.f / lstat[e];
#pragma unroll
  for (int nf = 0; nf < 4; ++nf) {
    const int col = h * HD + nf * 16 + lane16;
#pragma unroll
    for (int e = 0; e < 8; ++e) {
      const int row = b * SLEN + q0 + e + hi * 8;
      Ao[(size_t)row * DIM_ + col] = (__bf16)(o[nf][e] * inv[e]);
    }
  }
}

// ---------------------------------------------------------------------------
// Output projection: fp32 out = Ao * Wo + bo.
// ---------------------------------------------------------------------------
__global__ void __launch_bounds__(256)
oproj_gemm_kernel(const __bf16* __restrict__ Ao, const __bf16* __restrict__ Wot,
                  const float* __restrict__ bo, float* __restrict__ out) {
  const int lane = threadIdx.x & 31;
  const int wave = threadIdx.x >> 5;
  const int tile = blockIdx.x * 8 + wave;     // 128 * 12 tiles
  const int tn = tile % (DIM_ / 64);
  const int tm = tile / (DIM_ / 64);
  const int m0 = tm * 64, n0 = tn * 64;
  const int lane16 = lane & 15, hi = lane >> 4;

  v8f acc[4][4];
#pragma unroll
  for (int i = 0; i < 4; ++i)
#pragma unroll
    for (int j = 0; j < 4; ++j) acc[i][j] = vzero8();

  gemm_acc_64x64(Ao + (size_t)m0 * DIM_, Wot + (size_t)n0 * DIM_,
                 DIM_, DIM_, DIM_, acc, lane16, hi);

#pragma unroll
  for (int i = 0; i < 4; ++i)
#pragma unroll
    for (int j = 0; j < 4; ++j) {
      const int ncol = n0 + j * 16 + lane16;
      const float bias = bo[ncol];
#pragma unroll
      for (int e = 0; e < 8; ++e) {
        const int m = m0 + i * 16 + e + hi * 8;
        out[(size_t)m * DIM_ + ncol] = acc[i][j][e] + bias;
      }
    }
}

// ---------------------------------------------------------------------------
// Host launch
// ---------------------------------------------------------------------------
extern "C" void kernel_launch(void* const* d_in, const int* in_sizes, int n_in,
                              void* d_out, int out_size, void* d_ws, size_t ws_size,
                              hipStream_t stream) {
  const float* x    = (const float*)d_in[0];
  const float* mask = (const float*)d_in[1];
  const float* Wq   = (const float*)d_in[2];
  const float* bq   = (const float*)d_in[3];
  const float* Wk   = (const float*)d_in[4];
  const float* bk   = (const float*)d_in[5];
  const float* Wv   = (const float*)d_in[6];
  const float* bv   = (const float*)d_in[7];
  const float* Wo   = (const float*)d_in[8];
  const float* bo   = (const float*)d_in[9];
  float* out = (float*)d_out;

  // workspace carve-up (bf16 elements); total ~67.6 MB
  __bf16* xb  = (__bf16*)d_ws;
  __bf16* Wt  = xb  + (size_t)MROWS * DIM_;          //  6,291,456
  __bf16* Wot = Wt  + (size_t)NQKV  * DIM_;          // +1,769,472
  __bf16* Qb  = Wot + (size_t)DIM_  * DIM_;          // +  589,824
  __bf16* Kb  = Qb  + (size_t)MROWS * HD * NHEAD / NHEAD * 0 + (size_t)4 * NHEAD * SLEN * HD;  // see below
  // (Q/K/Vt/Ao are each B*H*S*HD = 6,291,456 elements)
  Kb = Qb + (size_t)4 * NHEAD * SLEN * HD;
  __bf16* Vt = Kb + (size_t)4 * NHEAD * SLEN * HD;
  __bf16* Ao = Vt + (size_t)4 * NHEAD * SLEN * HD;

  // 1. pack activations & weights to bf16 (weights transposed)
  pack_x_kernel<<<(MROWS * DIM_ / 4 + 255) / 256, 256, 0, stream>>>(
      x, xb, MROWS * DIM_);
  pack_wqkv_kernel<<<(NQKV * DIM_ + 255) / 256, 256, 0, stream>>>(Wq, Wk, Wv, Wt);
  pack_wo_kernel<<<(DIM_ * DIM_ + 255) / 256, 256, 0, stream>>>(Wo, Wot);

  // 2. fused QKV projection: [8192,768] x [768,2304]
  qkv_gemm_kernel<<<(128 * (NQKV / 64)) / 8, 256, 0, stream>>>(
      xb, Wt, bq, bk, bv, Qb, Kb, Vt);

  // 3. flash attention: 4*12*128 query tiles, 8 waves/block
  attn_kernel<<<(4 * NHEAD * (SLEN / 16)) / 8, 256, 0, stream>>>(
      Qb, Kb, Vt, mask, Ao);

  // 4. output projection -> fp32
  oproj_gemm_kernel<<<(128 * (DIM_ / 64)) / 8, 256, 0, stream>>>(
      Ao, Wot, bo, out);
}
